// GraphConstructor_71983651881194
// MI455X (gfx1250) — compile-verified, matrix operand order checked by reading, not run on hardware
//
#include <hip/hip_runtime.h>
#include <hip/hip_bf16.h>

// ---------------------------------------------------------------------------
// GraphConstructor (k-NN graph from pairwise exp(-L2/d) similarities)
//   x: [8192, 512] f32
//   out (flat f32): edge_index rows [N*K] | edge_index cols [N*K] |
//                   edge_weight [N*K]     | W [N, N]
// Strategy: G = X X^T via bf16 WMMA (X fits in L2: 8 MB as bf16), streaming
// per-row top-16 so the full sims matrix is never materialized; W is memset
// to zero and only N*K winners are scattered (saves a 512 MB round trip --
// the memset of W at 256 MB is the bandwidth floor on 23.3 TB/s HBM).
// ---------------------------------------------------------------------------

typedef __attribute__((ext_vector_type(16))) __bf16 v16bf;
typedef __attribute__((ext_vector_type(4)))  __bf16 v4bf;
typedef __attribute__((ext_vector_type(8)))  float  v8f;

#define N_NODES 8192
#define D_FEAT  512
#define KNN     16
#define BM      64      // rows per workgroup
#define BN      128     // cols per chunk
#define NK      (N_NODES * KNN)

// ---------------- prep: row squared norms + f32 -> bf16 conversion ----------
__global__ __launch_bounds__(128) void prep_kernel(const float* __restrict__ x,
                                                   __bf16* __restrict__ xb,
                                                   float* __restrict__ sq) {
    __shared__ float red[128];
    const int row = blockIdx.x;
    const int t   = threadIdx.x;                      // 128 threads * 4 = 512
    float4 v = ((const float4*)(x + (size_t)row * D_FEAT))[t];
    v4bf o;
    o[0] = (__bf16)v.x; o[1] = (__bf16)v.y; o[2] = (__bf16)v.z; o[3] = (__bf16)v.w;
    *(v4bf*)(xb + (size_t)row * D_FEAT + t * 4) = o;
    red[t] = v.x * v.x + v.y * v.y + v.z * v.z + v.w * v.w;
    __syncthreads();
    #pragma unroll
    for (int off = 64; off > 0; off >>= 1) {
        if (t < off) red[t] += red[t + off];
        __syncthreads();
    }
    if (t == 0) sq[row] = red[0];
}

// ---------------- sorted top-K insertion (fully unrolled -> registers) ------
__device__ __forceinline__ void topk_insert(float (&tv)[KNN], int (&tc)[KNN],
                                            float v, int c) {
    if (v > tv[KNN - 1]) {
        tv[KNN - 1] = v; tc[KNN - 1] = c;
        #pragma unroll
        for (int j = KNN - 1; j > 0; --j) {
            if (tv[j] > tv[j - 1]) {
                float fv = tv[j]; tv[j] = tv[j - 1]; tv[j - 1] = fv;
                int   ic = tc[j]; tc[j] = tc[j - 1]; tc[j - 1] = ic;
            }
        }
    }
}

union FragLd { v16bf v; uint4 q[2]; };

// ---------------- main fused GEMM + sims + streaming top-k ------------------
__global__ __launch_bounds__(256) void graph_kernel(const __bf16* __restrict__ xb,
                                                    const float* __restrict__ sq,
                                                    float* __restrict__ out) {
    __shared__ float s_tile[BM][BN + 4];   // 33792 B; merge buffers alias it

    const int tid     = threadIdx.x;
    const int lane    = tid & 31;
    const int wave    = tid >> 5;          // 8 waves (wave32)
    const int l16     = lane & 15;
    const int hi      = lane >> 4;         // which 16-lane half
    const int waveRow = wave >> 2;         // 0..1  -> 32-row strip
    const int waveCol = wave & 3;          // 0..3  -> 32-col strip
    const int blockRow = blockIdx.x * BM;
    const int rBase    = blockRow + waveRow * 32;

    // Hoist sq[] for this wave's 32 rows (layout matches C fragment rows).
    float sqm[2][8];
    #pragma unroll
    for (int mi = 0; mi < 2; ++mi)
        #pragma unroll
        for (int j = 0; j < 8; ++j)
            sqm[mi][j] = sq[rBase + mi * 16 + hi * 8 + j];

    // Per-thread partial top-K (thread t covers row t>>2, cols (t&3)::4).
    float tv[KNN]; int tc[KNN];
    #pragma unroll
    for (int j = 0; j < KNN; ++j) { tv[j] = -INFINITY; tc[j] = 0; }

    const int r   = tid >> 2;
    const int sub = tid & 3;

    for (int chunk = 0; chunk < N_NODES / BN; ++chunk) {
        const int cBase = chunk * BN + waveCol * 32;

        v8f acc[2][2] = {};
        #pragma unroll 4
        for (int k0 = 0; k0 < D_FEAT; k0 += 32) {
            // A fragments: rows of X. 16-bit A 16x32 layout:
            //   lanes 0-15: K=k0+0..7 (v0-3), k0+16..23 (v4-7); lanes 16-31: +8
            v16bf a[2], b[2];
            #pragma unroll
            for (int mi = 0; mi < 2; ++mi) {
                const __bf16* p = xb + (size_t)(rBase + mi * 16 + l16) * D_FEAT
                                     + k0 + hi * 8;
                FragLd f;
                f.q[0] = *(const uint4*)(p);
                f.q[1] = *(const uint4*)(p + 16);
                a[mi] = f.v;
            }
            // B fragments: B[k][n] = X[n][k] (Gram), symmetric lane layout.
            #pragma unroll
            for (int ni = 0; ni < 2; ++ni) {
                const __bf16* p = xb + (size_t)(cBase + ni * 16 + l16) * D_FEAT
                                     + k0 + hi * 8;
                FragLd f;
                f.q[0] = *(const uint4*)(p);
                f.q[1] = *(const uint4*)(p + 16);
                b[ni] = f.v;
            }
            #pragma unroll
            for (int mi = 0; mi < 2; ++mi)
                #pragma unroll
                for (int ni = 0; ni < 2; ++ni)
                    acc[mi][ni] = __builtin_amdgcn_wmma_f32_16x16x32_bf16(
                        false, a[mi], false, b[ni], (short)0, acc[mi][ni],
                        false, false);
        }

        // d2 = |a|^2 + |b|^2 - 2 a.b ; sim = exp(-sqrt(d2)/D) ; stage to LDS.
        #pragma unroll
        for (int ni = 0; ni < 2; ++ni) {
            const float sqn = sq[cBase + ni * 16 + l16];
            #pragma unroll
            for (int mi = 0; mi < 2; ++mi) {
                #pragma unroll
                for (int j = 0; j < 8; ++j) {
                    float d2 = sqm[mi][j] + sqn - 2.0f * acc[mi][ni][j];
                    d2 = fmaxf(d2, 0.0f);
                    float dist = (d2 > 0.0f) ? __fsqrt_rn(d2) : 0.0f;
                    float sim  = __expf(-dist * (1.0f / (float)D_FEAT));
                    s_tile[waveRow * 32 + mi * 16 + hi * 8 + j]
                          [waveCol * 32 + ni * 16 + l16] = sim;
                }
            }
        }
        __syncthreads();

        // Streaming top-K scan of this 64x128 tile: 4 threads per row.
        #pragma unroll 8
        for (int c = sub; c < BN; c += 4)
            topk_insert(tv, tc, s_tile[r][c], chunk * BN + c);
        __syncthreads();
    }

    // Merge the 4 partial lists per row through (reused) LDS.
    float* s_mv = &s_tile[0][0];              // 64*64 floats
    int*   s_mc = (int*)(&s_tile[0][0]) + BM * 64;
    #pragma unroll
    for (int j = 0; j < KNN; ++j) {
        s_mv[r * 64 + sub * KNN + j] = tv[j];
        s_mc[r * 64 + sub * KNN + j] = tc[j];
    }
    __syncthreads();

    if (tid < BM) {
        float fv[KNN]; int fc[KNN];
        #pragma unroll
        for (int j = 0; j < KNN; ++j) { fv[j] = -INFINITY; fc[j] = 0; }
        for (int i = 0; i < 64; ++i)
            topk_insert(fv, fc, s_mv[tid * 64 + i], s_mc[tid * 64 + i]);

        const int row = blockRow + tid;
        #pragma unroll
        for (int j = 0; j < KNN; ++j) {
            const int   col = fc[j];
            const float val = fv[j];
            out[(size_t)row * KNN + j]              = (float)row;   // edge rows
            out[(size_t)NK + row * KNN + j]         = (float)col;   // edge cols
            out[(size_t)2 * NK + row * KNN + j]     = val;          // weights
            out[(size_t)3 * NK + (size_t)row * N_NODES + col] = val; // W scatter
        }
    }
}

// ---------------------------------------------------------------------------
extern "C" void kernel_launch(void* const* d_in, const int* in_sizes, int n_in,
                              void* d_out, int out_size, void* d_ws, size_t ws_size,
                              hipStream_t stream) {
    (void)in_sizes; (void)n_in; (void)out_size; (void)ws_size;

    const float* x  = (const float*)d_in[0];
    __bf16*      xb = (__bf16*)d_ws;                          // 8 MB bf16 copy of X
    float*       sq = (float*)((char*)d_ws +
                               (size_t)N_NODES * D_FEAT * sizeof(__bf16)); // 32 KB
    float*       out = (float*)d_out;

    // Zero W region (only top-k entries get scattered later).
    hipMemsetAsync(out + (size_t)3 * NK, 0,
                   (size_t)N_NODES * N_NODES * sizeof(float), stream);

    prep_kernel<<<N_NODES, 128, 0, stream>>>(x, xb, sq);
    graph_kernel<<<N_NODES / BM, 256, 0, stream>>>(xb, sq, out);
}
